// CrossAttentionBlock_1486058684922
// MI455X (gfx1250) — matrix operand order repaired and checked by reading, not compiled, not run on hardware
//
#include <hip/hip_runtime.h>

// CDNA5 (gfx1250) fused cross-attention block.
// Stage 1: conv3x3 + RMSNorm + Wq proj   -> qp  (bf16, [B,S,64], scaled 0.25*log2e)
// Stage 2: RMSNorm + Wk proj, v cast     -> kp  (bf16, [B,S,64]), vb (bf16, [B,64,S])
//          + per-(batch,head) max ||k_row|| via atomicMax (softmax bound)
// Stage 3: two-pass flash attention (denominator / PV) with
//          v_wmma_f32_16x16x32_bf16. Softmax shift m = ||q_row||*kmax
//          (Cauchy-Schwarz bound; softmax is invariant to the shift, the bound
//          just guarantees exp2 args in [-inf, 0]). The shift rides in the
//          WMMA C operand; exponentials use raw v_exp_f32 (flush-tail is fine).

typedef __attribute__((ext_vector_type(16))) __bf16 v16bf;
typedef __attribute__((ext_vector_type(8)))  __bf16 v8bf;
typedef __attribute__((ext_vector_type(8)))  float  v8f;

#define B_   8
#define C_   64
#define H_   48
#define S_   2304           // 48*48
#define NT_  144            // S_/16 key tiles (pass B)
#define NT2_ 72             // S_/32 key tiles (pass C)
#define EPS_ 1.1920929e-07f
#define QSCALE_ (0.25f * 1.44269504088896340736f)   // 1/sqrt(hd) * log2(e)

static __device__ __forceinline__ __bf16 f2bf(float f) {
  // round-to-nearest-even f32 -> bf16
  unsigned u = __builtin_bit_cast(unsigned, f);
  u += 0x7fffu + ((u >> 16) & 1u);
  unsigned short h = (unsigned short)(u >> 16);
  return __builtin_bit_cast(__bf16, h);
}

// ---------------------------------------------------------------------------
// Kernel 1: per (b,y) row: 3x3 SAME conv on q, RMSNorm over C, project with wq
// ---------------------------------------------------------------------------
__global__ __launch_bounds__(256) void k_conv_rms_projq(
    const float* __restrict__ q, const float* __restrict__ cw,
    const float* __restrict__ nqw, const float* __restrict__ wq,
    const float* __restrict__ bq, __bf16* __restrict__ qp) {
  __shared__ float in_s[3][C_][H_];   // 3 input rows, all channels
  __shared__ float qc_s[H_][C_];      // conv output [x][oc]
  __shared__ float scale_s[H_];

  const int b = blockIdx.x / H_;
  const int y = blockIdx.x % H_;
  const int tid = threadIdx.x;

  for (int idx = tid; idx < 3 * C_ * H_; idx += 256) {
    int ky = idx / (C_ * H_);
    int rem = idx % (C_ * H_);
    int c = rem / H_, x = rem % H_;
    int yy = y + ky - 1;
    float v = 0.f;
    if (yy >= 0 && yy < H_) v = q[((b * C_ + c) * H_ + yy) * H_ + x];
    in_s[ky][c][x] = v;
  }
  __syncthreads();

  for (int idx = tid; idx < H_ * C_; idx += 256) {
    int x = idx / C_, oc = idx % C_;
    float acc = 0.f;
    for (int c = 0; c < C_; ++c) {
      const float* wp = cw + (oc * C_ + c) * 9;
      #pragma unroll
      for (int ky = 0; ky < 3; ++ky) {
        #pragma unroll
        for (int kx = 0; kx < 3; ++kx) {
          int xx = x + kx - 1;
          if (xx >= 0 && xx < H_) acc += in_s[ky][c][xx] * wp[ky * 3 + kx];
        }
      }
    }
    qc_s[x][oc] = acc;
  }
  __syncthreads();

  if (tid < H_) {
    float ss = 0.f;
    for (int c = 0; c < C_; ++c) { float v = qc_s[tid][c]; ss += v * v; }
    scale_s[tid] = rsqrtf(ss * (1.0f / C_) + EPS_);
  }
  __syncthreads();

  for (int idx = tid; idx < H_ * C_; idx += 256) {
    int x = idx / C_, e = idx % C_;
    float s = scale_s[x];
    float acc = bq[e];
    const float* wr = wq + e * C_;
    for (int c = 0; c < C_; ++c) acc += qc_s[x][c] * s * nqw[c] * wr[c];
    // fold torch's 1/sqrt(head_dim) AND log2(e) (exp->exp2 in softmax) into q
    qp[(b * S_ + y * H_ + x) * C_ + e] = f2bf(acc * QSCALE_);
  }
}

// ---------------------------------------------------------------------------
// Kernel 2: RMSNorm + Wk proj for k -> kp [B,S,64]; raw v cast -> vb [B,64,S];
//           per-(b,head) max row norm of kp -> kmax[b*4+h] (int-atomicMax).
// ---------------------------------------------------------------------------
__global__ __launch_bounds__(256) void k_rms_projk_v(
    const float* __restrict__ k, const float* __restrict__ v,
    const float* __restrict__ nkw, const float* __restrict__ wk,
    const float* __restrict__ bk, __bf16* __restrict__ kp,
    __bf16* __restrict__ vb, float* __restrict__ kmax) {
  __shared__ float kt[64][C_ + 1];    // [pos][ch] input
  __shared__ float kq[64][C_ + 1];    // [pos][e]  projected (f32)
  __shared__ float scale_s[64];
  const int b = blockIdx.x / 36;
  const int s0 = (blockIdx.x % 36) * 64;
  const int tid = threadIdx.x;

  for (int idx = tid; idx < 64 * C_; idx += 256) {
    int c = idx / 64, s = idx % 64;
    int g = (b * C_ + c) * S_ + s0 + s;
    kt[s][c] = k[g];
    vb[g] = f2bf(v[g]);               // keep [B,C,S] layout (K-major for PV WMMA)
  }
  __syncthreads();

  if (tid < 64) {
    float ss = 0.f;
    for (int c = 0; c < C_; ++c) { float x = kt[tid][c]; ss += x * x; }
    scale_s[tid] = rsqrtf(ss * (1.0f / C_) + EPS_);
  }
  __syncthreads();

  for (int idx = tid; idx < 64 * C_; idx += 256) {
    int s = idx / C_, e = idx % C_;
    float acc = bk[e];
    float sc = scale_s[s];
    const float* wr = wk + e * C_;
    for (int c = 0; c < C_; ++c) acc += kt[s][c] * sc * nkw[c] * wr[c];
    kq[s][e] = acc;
    kp[(b * S_ + s0 + s) * C_ + e] = f2bf(acc);
  }
  __syncthreads();

  // 256 threads = 64 positions x 4 heads: per-head row norm, global max
  {
    int s = tid >> 2, h = tid & 3;
    float ss = 0.f;
    #pragma unroll
    for (int t = 0; t < 16; ++t) { float x = kq[s][h * 16 + t]; ss += x * x; }
    float nrm = sqrtf(ss);
    atomicMax((int*)(kmax + b * 4 + h), __float_as_int(nrm)); // non-neg floats
  }
}

// ---------------------------------------------------------------------------
// Kernel 3: two-pass flash attention, head-avg softmax, PV with raw v.
// One wave = one 16-row query tile. 8 waves per block.
// B-fragment elements 8-15 (K=16..31) multiply the zero-padded upper half of the
// Q A-fragments for score WMMAs, so they can load arbitrary finite data ->
// all loads unconditional. Softmax shift rides in the WMMA C operand.
// ---------------------------------------------------------------------------
__global__ __launch_bounds__(256) void k_flash_attn(
    const __bf16* __restrict__ qp, const __bf16* __restrict__ kp,
    const __bf16* __restrict__ vb, const float* __restrict__ kmax,
    float* __restrict__ out) {
  __shared__ __align__(32) __bf16 p_s[8][16][32];   // per-wave P staging (16x32)

  const int wave = threadIdx.x >> 5;
  const int lane = threadIdx.x & 31;
  const int half = lane >> 4;        // 0: rows 0-7 / K 0-7 ; 1: rows 8-15 / K 8-15
  const int n    = lane & 15;
  const int b    = blockIdx.x / 18;
  const int i0   = ((blockIdx.x % 18) * 8 + wave) * 16;

  const v16bf z16 = {};
  const __bf16* kbase = kp + (size_t)b * S_ * C_;

  // Q A-fragments (K 0..15 = head dims, K 16..31 zero-padded) + row norms.
  // A layout: M = lane&15; lanes<16 hold K 0-7 (elems 0-7), lanes>=16 K 8-15.
  v16bf aq[4];
  float mrow[4];                      // per-lane: softmax shift for row n
  #pragma unroll
  for (int h = 0; h < 4; ++h) {
    v8bf lo = *(const v8bf*)(qp + (size_t)(b * S_ + i0 + n) * C_ + h * 16 + half * 8);
    v16bf a = z16;
    float qss = 0.f;
    #pragma unroll
    for (int t = 0; t < 8; ++t) {
      a[t] = lo[t];
      float x = (float)lo[t];
      qss += x * x;
    }
    aq[h] = a;
    qss += __shfl_xor(qss, 16, 32);   // join the two K-halves of row n
    // Cauchy-Schwarz upper bound on this row's scores (log2 domain)
    mrow[h] = sqrtf(qss) * kmax[b * 4 + h];
  }

  // Redistribute: D-fragment row r (+half*8) lives in source lane (r+half*8).
  // nm[h][r] = -m for that row, uniform across lanes -> WMMA C operand.
  v8f nm[4];
  #pragma unroll
  for (int h = 0; h < 4; ++h)
    #pragma unroll
    for (int r = 0; r < 8; ++r)
      nm[h][r] = -__shfl(mrow[h], r + half * 8, 32);

  // ---- pass B: per-lane denominator; merge across lanes once ----
  float l[4][8];
  #pragma unroll
  for (int h = 0; h < 4; ++h)
    #pragma unroll
    for (int r = 0; r < 8; ++r) l[h][r] = 0.f;

  for (int jt = 0; jt < NT_; ++jt) {
    const int j0 = jt * 16;
    {
      int jp = j0 + 128 + n;          // stream prefetch ahead on the K rows
      if (jp < S_) __builtin_prefetch(kbase + (size_t)jp * C_, 0, 1);
    }
    #pragma unroll
    for (int h = 0; h < 4; ++h) {
      v8bf kv = *(const v8bf*)(kbase + (size_t)(j0 + n) * C_ + h * 16 + half * 8);
      v16bf bkf = __builtin_shufflevector(kv, kv, 0, 1, 2, 3, 4, 5, 6, 7, 0, 1, 2, 3, 4, 5, 6, 7);
      // D = q.k - m  (shift applied by the XDL via C)
      v8f sc = __builtin_amdgcn_wmma_f32_16x16x32_bf16(
          false, aq[h], false, bkf, (short)0, nm[h], false, false);
      #pragma unroll
      for (int r = 0; r < 8; ++r)
        l[h][r] += __builtin_amdgcn_exp2f(sc[r]);   // raw v_exp_f32, args <= 0
    }
  }
  #pragma unroll
  for (int h = 0; h < 4; ++h)
    #pragma unroll
    for (int r = 0; r < 8; ++r) {
      float s = l[h][r];
      s += __shfl_xor(s, 1, 32);
      s += __shfl_xor(s, 2, 32);
      s += __shfl_xor(s, 4, 32);
      s += __shfl_xor(s, 8, 32);
      l[h][r] = 0.25f / s;            // fold head-average into normalizer
    }

  v8f oacc[4];
  #pragma unroll
  for (int c = 0; c < 4; ++c) oacc[c] = (v8f){};

  // ---- pass C: rebuild probs for 32 keys/iter, average heads, P @ V ----
  for (int jt = 0; jt < NT2_; ++jt) {
    const int j0 = jt * 32;
    float pacc0[8], pacc1[8];
    #pragma unroll
    for (int r = 0; r < 8; ++r) { pacc0[r] = 0.f; pacc1[r] = 0.f; }

    #pragma unroll
    for (int h = 0; h < 4; ++h) {
      v8bf kv0 = *(const v8bf*)(kbase + (size_t)(j0 + n) * C_ + h * 16 + half * 8);
      v8bf kv1 = *(const v8bf*)(kbase + (size_t)(j0 + 16 + n) * C_ + h * 16 + half * 8);
      v16bf bk0 = __builtin_shufflevector(kv0, kv0, 0, 1, 2, 3, 4, 5, 6, 7, 0, 1, 2, 3, 4, 5, 6, 7);
      v16bf bk1 = __builtin_shufflevector(kv1, kv1, 0, 1, 2, 3, 4, 5, 6, 7, 0, 1, 2, 3, 4, 5, 6, 7);
      v8f s0 = __builtin_amdgcn_wmma_f32_16x16x32_bf16(
          false, aq[h], false, bk0, (short)0, nm[h], false, false);
      v8f s1 = __builtin_amdgcn_wmma_f32_16x16x32_bf16(
          false, aq[h], false, bk1, (short)0, nm[h], false, false);
      #pragma unroll
      for (int r = 0; r < 8; ++r) {
        pacc0[r] += __builtin_amdgcn_exp2f(s0[r]) * l[h][r];
        pacc1[r] += __builtin_amdgcn_exp2f(s1[r]) * l[h][r];
      }
    }

    // scores D layout (lane=col) -> 16x32 A layout (lane=row) via LDS
    #pragma unroll
    for (int r = 0; r < 8; ++r) {
      p_s[wave][r + half * 8][n]      = f2bf(pacc0[r]);
      p_s[wave][r + half * 8][16 + n] = f2bf(pacc1[r]);
    }
    __builtin_amdgcn_wave_barrier();   // LDS in-order per wave; fence scheduler

    // P A-fragment: all 32 K slots real.
    v16bf ap;
    {
      v8bf lo = *(const v8bf*)&p_s[wave][n][half * 16];
      v8bf hi = *(const v8bf*)&p_s[wave][n][half * 16 + 8];
      #pragma unroll
      for (int t = 0; t < 8; ++t) { ap[t] = lo[t]; ap[8 + t] = hi[t]; }
    }
    __builtin_amdgcn_wave_barrier();

    #pragma unroll
    for (int c = 0; c < 4; ++c) {
      // V B-fragment, full K=32: lanes<16 K=j0..j0+15, lanes>=16 K=j0+16..j0+31
      v16bf bvf = *(const v16bf*)(vb + (size_t)(b * C_ + c * 16 + n) * S_
                                  + j0 + half * 16);
      oacc[c] = __builtin_amdgcn_wmma_f32_16x16x32_bf16(
          false, ap, false, bvf, (short)0, oacc[c], false, false);
    }
  }

  // store: D layout — row = r + half*8, col = c*16 + n
  #pragma unroll
  for (int c = 0; c < 4; ++c)
    #pragma unroll
    for (int r = 0; r < 8; ++r)
      out[(size_t)(b * S_ + i0 + r + half * 8) * C_ + c * 16 + n] = oacc[c][r];
}

// ---------------------------------------------------------------------------
extern "C" void kernel_launch(void* const* d_in, const int* in_sizes, int n_in,
                              void* d_out, int out_size, void* d_ws, size_t ws_size,
                              hipStream_t stream) {
  const float* q   = (const float*)d_in[0];
  const float* k   = (const float*)d_in[1];
  const float* v   = (const float*)d_in[2];
  const float* cw  = (const float*)d_in[3];
  const float* nqw = (const float*)d_in[4];
  const float* nkw = (const float*)d_in[5];
  const float* wq  = (const float*)d_in[6];
  const float* bq  = (const float*)d_in[7];
  const float* wk  = (const float*)d_in[8];
  const float* bk  = (const float*)d_in[9];
  float* out = (float*)d_out;

  const size_t nElem = (size_t)B_ * S_ * C_;        // 1,179,648
  __bf16* qp = (__bf16*)d_ws;                       // [B,S,64]
  __bf16* kp = qp + nElem;                          // [B,S,64]
  __bf16* vb = kp + nElem;                          // [B,64,S]
  float* kmx = (float*)(vb + nElem);                // [B,4] max ||k_row|| per head

  hipMemsetAsync(kmx, 0, B_ * 4 * sizeof(float), stream);  // capture-legal
  k_conv_rms_projq<<<B_ * H_, 256, 0, stream>>>(q, cw, nqw, wq, bq, qp);
  k_rms_projk_v<<<B_ * 36, 256, 0, stream>>>(k, v, nkw, wk, bk, kp, vb, kmx);
  k_flash_attn<<<B_ * 18, 256, 0, stream>>>(qp, kp, vb, kmx, out);
}
